// TransitionDown_15710990369320
// MI455X (gfx1250) — matrix operand order repaired and checked by reading, not compiled, not run on hardware
//
#include <hip/hip_runtime.h>
#include <hip/hip_bf16.h>

typedef __attribute__((ext_vector_type(16))) _Float16 v16h;
typedef __attribute__((ext_vector_type(8)))  _Float16 v8h;
typedef __attribute__((ext_vector_type(8)))  float    v8f;

#define B_    16
#define N_    4096
#define D_    64
#define NP    1024
#define NS    16
#define DCH   128
#define K0PAD 96
#define MTOT  (B_*NP*NS)   /* 262144 GEMM rows */
#define EPSB  1e-5f

// ---------------- workspace layout (256-B aligned offsets) ----------------
#define OFF_FPS     0u            /* int   16*1024            =   65536 B */
#define OFF_NEWXYZ  65536u        /* float 16*1024*3          =  196608 B */
#define OFF_KNN     262144u       /* int   16*1024*16         = 1048576 B */
#define OFF_W0      1310720u      /* f16 swizzled 8*3*32*16   =   24576 B */
#define OFF_W1      1335296u      /* f16 swizzled 8*4*32*16   =   32768 B */
#define OFF_MV0     1368064u      /* float 256                            */
#define OFF_MV1     1369088u      /* float 256                            */
#define OFF_PART    1370112u      /* float 256*256            =  262144 B */
#define OFF_CLSC    1632256u      /* float 16*16*128          =  131072 B */
#define OFF_A0      1763328u      /* f16   262144*96          = 50331648 B*/
#define OFF_H       52094976u     /* float 262144*128         =134217728 B (reused h0->h1) */
#define OFF_A1      186312704u    /* f16   262144*128         = 67108864 B*/

// ======================= 1. Farthest point sampling =======================
__global__ void fps_kernel(const float* __restrict__ xyz, int* __restrict__ fps_idx) {
    const int b = blockIdx.x, tid = threadIdx.x;           // 256 threads
    __shared__ float sv[256];
    __shared__ int   si[256];
    __shared__ float cen[3];
    float px[16], py[16], pz[16], dmin[16];
    const float* base = xyz + (size_t)b * N_ * 3;
#pragma unroll
    for (int i = 0; i < 16; ++i) {
        int n = i * 256 + tid;
        px[i] = base[n*3+0]; py[i] = base[n*3+1]; pz[i] = base[n*3+2];
        dmin[i] = 1e10f;
    }
    int far = 0;
    for (int it = 0; it < NP; ++it) {
        if (tid == 0) {
            fps_idx[b*NP + it] = far;
            cen[0] = base[far*3+0]; cen[1] = base[far*3+1]; cen[2] = base[far*3+2];
        }
        __syncthreads();
        float cx = cen[0], cy = cen[1], cz = cen[2];
        float best = -1.0f; int bidx = 0;
#pragma unroll
        for (int i = 0; i < 16; ++i) {
            float dx = px[i]-cx, dy = py[i]-cy, dz = pz[i]-cz;
            float d  = dx*dx + dy*dy + dz*dz;
            dmin[i]  = fminf(dmin[i], d);
            if (dmin[i] > best) { best = dmin[i]; bidx = i*256 + tid; }
        }
        sv[tid] = best; si[tid] = bidx;
        __syncthreads();
        for (int s = 128; s > 0; s >>= 1) {
            if (tid < s) {
                if (sv[tid+s] > sv[tid] ||
                    (sv[tid+s] == sv[tid] && si[tid+s] < si[tid])) {
                    sv[tid] = sv[tid+s]; si[tid] = si[tid+s];
                }
            }
            __syncthreads();
        }
        far = si[0];
    }
}

// ======================= 2. gather new_xyz ================================
__global__ void gather_new_xyz(const float* __restrict__ xyz,
                               const int* __restrict__ fps_idx,
                               float* __restrict__ new_xyz) {
    int i = blockIdx.x * blockDim.x + threadIdx.x;          // B*NP
    if (i >= B_*NP) return;
    int b = i >> 10;
    int idx = fps_idx[i];
    const float* s = xyz + ((size_t)b*N_ + idx)*3;
    new_xyz[i*3+0] = s[0]; new_xyz[i*3+1] = s[1]; new_xyz[i*3+2] = s[2];
}

// ======================= 3. 16-NN (register insertion net) ================
__global__ void knn_kernel(const float* __restrict__ xyz,
                           const float* __restrict__ new_xyz,
                           int* __restrict__ knn_idx) {
    int gid = blockIdx.x * blockDim.x + threadIdx.x;        // B*NP queries
    if (gid >= B_*NP) return;
    int b = gid >> 10;
    float qx = new_xyz[gid*3+0], qy = new_xyz[gid*3+1], qz = new_xyz[gid*3+2];
    const float* base = xyz + (size_t)b * N_ * 3;
    float bd[NS]; int bi[NS];
#pragma unroll
    for (int j = 0; j < NS; ++j) { bd[j] = 3.4e38f; bi[j] = 0; }
    for (int n = 0; n < N_; ++n) {
        float dx = base[n*3+0]-qx, dy = base[n*3+1]-qy, dz = base[n*3+2]-qz;
        float d  = dx*dx + dy*dy + dz*dz;
        if (d < bd[NS-1]) {
#pragma unroll
            for (int j = NS-1; j >= 1; --j) {
                if (bd[j] > d) {
                    if (bd[j-1] > d) { bd[j] = bd[j-1]; bi[j] = bi[j-1]; }
                    else             { bd[j] = d;       bi[j] = n;       }
                }
            }
            if (bd[0] > d) { bd[0] = d; bi[0] = n; }
        }
    }
#pragma unroll
    for (int j = 0; j < NS; ++j) knn_idx[(size_t)gid*NS + j] = bi[j];
}

// ======================= 4. build A0 (f16, K padded to 96) ================
__global__ void build_a0(const float* __restrict__ xyz,
                         const float* __restrict__ points,
                         const float* __restrict__ new_xyz,
                         const int* __restrict__ knn_idx,
                         _Float16* __restrict__ A0) {
    int m = blockIdx.x * blockDim.x + threadIdx.x;          // MTOT rows
    if (m >= MTOT) return;
    int n  = m & (NS-1);
    int bk = m >> 4;
    int b  = bk >> 10;
    int idx = knn_idx[(size_t)bk*NS + n];
    const float* p = xyz + ((size_t)b*N_ + idx)*3;
    const float* q = new_xyz + (size_t)bk*3;
    _Float16* row = A0 + (size_t)m * K0PAD;
    row[0] = (_Float16)(p[0]-q[0]);
    row[1] = (_Float16)(p[1]-q[1]);
    row[2] = (_Float16)(p[2]-q[2]);
    const float* f = points + ((size_t)b*N_ + idx)*D_;
#pragma unroll 8
    for (int c = 0; c < D_; ++c) row[3+c] = (_Float16)f[c];
#pragma unroll
    for (int c = 3+D_; c < K0PAD; ++c) row[c] = (_Float16)0.f;
}

// ============ 5. weight prep: swizzle B into fragment-ready layout ========
// flat index = ((ntile*KTILES + kk)*32 + lane)*16 + j
// lane element j = B[kk*32 + (lane>=16 ? 16:0) + j][ntile*16 + (lane&15)]
__global__ void prep_w(const float* __restrict__ w0, const float* __restrict__ w1,
                       _Float16* __restrict__ W0s, _Float16* __restrict__ W1s) {
    int i = blockIdx.x * blockDim.x + threadIdx.x;          // 16384 threads
    {   // conv1 swizzle (KTILES = 4): 8*4*32*16 = 16384 halves
        int j = i & 15, lane = (i >> 4) & 31;
        int kk = (i >> 9) & 3, ntile = i >> 11;
        int col  = ntile*16 + (lane & 15);
        int krow = kk*32 + ((lane >= 16) ? 16 : 0) + j;
        W1s[i] = (_Float16)w1[col*DCH + krow];              // B[k][n] = w1[n*128+k]
    }
    if (i < 12288) {  // conv0 swizzle (KTILES = 3): 8*3*32*16 = 12288 halves
        int j = i & 15, lane = (i >> 4) & 31;
        int kidx = i >> 9;
        int kk = kidx % 3, ntile = kidx / 3;
        int col  = ntile*16 + (lane & 15);
        int krow = kk*32 + ((lane >= 16) ? 16 : 0) + j;
        W0s[i] = (krow < 67) ? (_Float16)w0[col*67 + krow] : (_Float16)0.f;
    }
}

// ======================= 6. WMMA GEMM: Out = A(MxK) * B(KxN=128) + bias ===
// block = 256 threads = 8 waves; blockIdx.x = 16-row M tile; wave w -> cols [16w,16w+16)
// Swizzled B staged to LDS via gfx1250 async global->LDS copies (ASYNCcnt).
template<int KTILES, int LDA>
__global__ void gemm_wmma(const _Float16* __restrict__ A,
                          const _Float16* __restrict__ Bsw,
                          const float* __restrict__ bias,
                          float* __restrict__ Out) {
    __shared__ __align__(32) _Float16 Bs[KTILES * 8 * 512];  // 8 ntiles*KTILES*32 lanes*16

    // ---- cooperative async copy of the whole swizzled B into LDS ----
    const int nchunks = KTILES * 512;                       // 16-B chunks
    {
        unsigned ldsbase = (unsigned)(uintptr_t)(&Bs[0]);
        for (int c = threadIdx.x; c < nchunks; c += 256) {
            unsigned      loff = ldsbase + (unsigned)c * 16u;
            unsigned long long gaddr =
                (unsigned long long)(uintptr_t)((const char*)Bsw + (size_t)c * 16);
            asm volatile("global_load_async_to_lds_b128 %0, %1, off"
                         :: "v"(loff), "v"(gaddr) : "memory");
        }
        asm volatile("s_wait_asynccnt 0x0" ::: "memory");
    }
    __syncthreads();

    const int wave  = threadIdx.x >> 5;
    const int lane  = threadIdx.x & 31;
    const int lhalf = lane & 15;
    const bool hi   = lane >= 16;
    const int row   = blockIdx.x * 16 + lhalf;   // A row for this lane's fragment
    const int col   = wave * 16 + lhalf;         // B/D column for this lane
    v8f c = {};
#pragma unroll
    for (int kk = 0; kk < KTILES; ++kk) {
        // ---- A fragment: lanes 0-15 K{0..15}, lanes 16-31 K{16..31}
        const v8h* ap = (const v8h*)(A + (size_t)row * LDA + kk*32 + (hi ? 16 : 0));
        v8h a0 = ap[0];            // K j..j+7
        v8h a1 = ap[1];            // +8 halves
        // ---- B fragment: contiguous 32 B per lane from LDS (ds_load_b128 x2)
        const v8h* bp = (const v8h*)(Bs + (((size_t)wave*KTILES + kk)*32 + lane)*16);
        v8h b0 = bp[0];
        v8h b1 = bp[1];
        v16h a, bf;
#pragma unroll
        for (int j = 0; j < 8; ++j) {
            a[j] = a0[j];  a[8+j] = a1[j];
            bf[j] = b0[j]; bf[8+j] = b1[j];
        }
        c = __builtin_amdgcn_wmma_f32_16x16x32_f16(false, a, false, bf,
                                                   (short)0, c, false, false);
    }
    const float bv = bias[col];
    const int rbase = blockIdx.x * 16 + (hi ? 8 : 0);
#pragma unroll
    for (int r = 0; r < 8; ++r)
        Out[(size_t)(rbase + r) * DCH + col] = c[r] + bv;
}

// ======================= 7. BN stats (two stage) ==========================
__global__ void stats_partial(const float* __restrict__ X,
                              float* __restrict__ partial, long total) {
    __shared__ float sacc[256];                  // [0..127] sum, [128..255] sumsq
    int tid = threadIdx.x;
    sacc[tid] = 0.f;
    __syncthreads();
    long stride = (long)gridDim.x * blockDim.x;
    for (long e = (long)blockIdx.x * blockDim.x + tid; e < total; e += stride) {
        float v = X[e];
        int ch = (int)(e & (DCH-1));
        atomicAdd(&sacc[ch], v);
        atomicAdd(&sacc[DCH+ch], v*v);
    }
    __syncthreads();
    partial[(size_t)blockIdx.x * 256 + tid] = sacc[tid];
}

__global__ void stats_final(const float* __restrict__ partial,
                            float* __restrict__ mv, int nblocks, float count) {
    int ch = threadIdx.x;                        // 128 threads
    float s = 0.f, q = 0.f;
    for (int i = 0; i < nblocks; ++i) {
        s += partial[(size_t)i*256 + ch];
        q += partial[(size_t)i*256 + DCH + ch];
    }
    float mean = s / count;
    mv[ch]     = mean;
    mv[DCH+ch] = q / count - mean*mean;
}

// ======================= 8. BN+ReLU -> f16 (feeds conv1) ==================
__global__ void bn_relu_h16(const float* __restrict__ X, const float* __restrict__ mv,
                            const float* __restrict__ g, const float* __restrict__ bt,
                            _Float16* __restrict__ Y) {
    size_t i = (size_t)blockIdx.x * blockDim.x + threadIdx.x;  // MTOT*DCH
    int ch = (int)(i & (DCH-1));
    float m = mv[ch], v = mv[DCH+ch];
    float y = (X[i]-m) * rsqrtf(v + EPSB) * g[ch] + bt[ch];
    Y[i] = (_Float16)fmaxf(y, 0.f);
}

// ======================= 9. BN+ReLU + max over 16 samples =================
__global__ void bn_relu_max(const float* __restrict__ H, const float* __restrict__ mv,
                            const float* __restrict__ g, const float* __restrict__ bt,
                            float* __restrict__ pts_out) {
    int i = blockIdx.x * blockDim.x + threadIdx.x;          // B*NP*DCH
    if (i >= B_*NP*DCH) return;
    int ch = i & (DCH-1);
    int bk = i >> 7;
    int b  = bk >> 10;
    int k  = bk & (NP-1);
    float m = mv[ch];
    float inv = rsqrtf(mv[DCH+ch] + EPSB) * g[ch];
    float bb = bt[ch];
    float mx = -3.4e38f;
#pragma unroll
    for (int n = 0; n < NS; ++n) {
        float x = H[((size_t)bk*NS + n)*DCH + ch];
        float y = fmaxf((x - m)*inv + bb, 0.f);
        mx = fmaxf(mx, y);
    }
    pts_out[((size_t)b*(NP+16) + k)*DCH + ch] = mx;
}

// ======================= 10. class branch =================================
__global__ void cls_fc1(const float* __restrict__ points, const float* __restrict__ w1,
                        const float* __restrict__ b1, float* __restrict__ C) {
    int i = blockIdx.x * blockDim.x + threadIdx.x;          // 16*16*128
    if (i >= B_*16*DCH) return;
    int ch = i & (DCH-1);
    int r  = i >> 7;
    int b  = r >> 4, p = r & 15;
    const float* x = points + ((size_t)b*N_ + p)*D_;
    float acc = b1[ch];
#pragma unroll 8
    for (int k = 0; k < D_; ++k) acc += x[k] * w1[ch*D_ + k];
    C[i] = fmaxf(acc, 0.f);
}

__global__ void cls_fc2(const float* __restrict__ C, const float* __restrict__ w2,
                        const float* __restrict__ b2, float* __restrict__ pts_out) {
    int i = blockIdx.x * blockDim.x + threadIdx.x;          // 16*16*128
    if (i >= B_*16*DCH) return;
    int ch = i & (DCH-1);
    int r  = i >> 7;
    int b  = r >> 4, p = r & 15;
    const float* x = C + (size_t)r * DCH;
    float acc = b2[ch];
#pragma unroll 8
    for (int k = 0; k < DCH; ++k) acc += x[k] * w2[ch*DCH + k];
    pts_out[((size_t)b*(NP+16) + NP + p)*DCH + ch] = acc;
}

// ======================= 11. xyz_out ======================================
__global__ void write_xyz(const float* __restrict__ xyz, const float* __restrict__ new_xyz,
                          float* __restrict__ xyz_out) {
    int i = blockIdx.x * blockDim.x + threadIdx.x;          // 16*1040*3 = 49920
    if (i >= B_*(NP+16)*3) return;
    int c = i % 3;
    int r = i / 3;
    int b = r / (NP+16), k = r % (NP+16);
    float v = (k < NP) ? new_xyz[((size_t)b*NP + k)*3 + c]
                       : xyz[((size_t)b*N_ + (k-NP))*3 + c];
    xyz_out[i] = v;
}

// =========================================================================
extern "C" void kernel_launch(void* const* d_in, const int* in_sizes, int n_in,
                              void* d_out, int out_size, void* d_ws, size_t ws_size,
                              hipStream_t stream) {
    const float* xyz     = (const float*)d_in[0];
    const float* points  = (const float*)d_in[1];
    const float* conv_w0 = (const float*)d_in[2];
    const float* conv_b0 = (const float*)d_in[3];
    const float* bn_g0   = (const float*)d_in[4];
    const float* bn_b0   = (const float*)d_in[5];
    const float* conv_w1 = (const float*)d_in[6];
    const float* conv_b1 = (const float*)d_in[7];
    const float* bn_g1   = (const float*)d_in[8];
    const float* bn_b1   = (const float*)d_in[9];
    const float* fc_w1   = (const float*)d_in[10];
    const float* fc_b1   = (const float*)d_in[11];
    const float* fc_w2   = (const float*)d_in[12];
    const float* fc_b2   = (const float*)d_in[13];

    char* ws = (char*)d_ws;
    int*      fps_idx = (int*)     (ws + OFF_FPS);
    float*    new_xyz = (float*)   (ws + OFF_NEWXYZ);
    int*      knn_idx = (int*)     (ws + OFF_KNN);
    _Float16* W0s     = (_Float16*)(ws + OFF_W0);
    _Float16* W1s     = (_Float16*)(ws + OFF_W1);
    float*    mv0     = (float*)   (ws + OFF_MV0);
    float*    mv1     = (float*)   (ws + OFF_MV1);
    float*    part    = (float*)   (ws + OFF_PART);
    float*    clsC    = (float*)   (ws + OFF_CLSC);
    _Float16* A0      = (_Float16*)(ws + OFF_A0);
    float*    H       = (float*)   (ws + OFF_H);      // conv0 out, then conv1 out
    _Float16* A1      = (_Float16*)(ws + OFF_A1);

    float* xyz_out = (float*)d_out;                    // 16*1040*3 floats
    float* pts_out = xyz_out + B_*(NP+16)*3;           // 16*1040*128 floats

    const long total_elems = (long)MTOT * DCH;         // 33,554,432

    fps_kernel      <<<B_, 256, 0, stream>>>(xyz, fps_idx);
    gather_new_xyz  <<<(B_*NP)/256, 256, 0, stream>>>(xyz, fps_idx, new_xyz);
    knn_kernel      <<<(B_*NP)/256, 256, 0, stream>>>(xyz, new_xyz, knn_idx);
    build_a0        <<<MTOT/256, 256, 0, stream>>>(xyz, points, new_xyz, knn_idx, A0);
    prep_w          <<<(DCH*DCH)/256, 256, 0, stream>>>(conv_w0, conv_w1, W0s, W1s);

    // conv0: 262144 x 96 (f16) * 96 x 128 (f16) -> f32 + bias
    gemm_wmma<3, K0PAD><<<MTOT/16, 256, 0, stream>>>(A0, W0s, conv_b0, H);
    stats_partial   <<<256, 256, 0, stream>>>(H, part, total_elems);
    stats_final     <<<1, 128, 0, stream>>>(part, mv0, 256, (float)total_elems);
    bn_relu_h16     <<<(unsigned)(total_elems/256), 256, 0, stream>>>(H, mv0, bn_g0, bn_b0, A1);

    // conv1: 262144 x 128 (f16) * 128 x 128 (f16) -> f32 + bias (H reused)
    gemm_wmma<4, DCH><<<MTOT/16, 256, 0, stream>>>(A1, W1s, conv_b1, H);
    stats_partial   <<<256, 256, 0, stream>>>(H, part, total_elems);
    stats_final     <<<1, 128, 0, stream>>>(part, mv1, 256, (float)total_elems);
    bn_relu_max     <<<(B_*NP*DCH)/256, 256, 0, stream>>>(H, mv1, bn_g1, bn_b1, pts_out);

    // class branch + xyz concat
    cls_fc1         <<<(B_*16*DCH)/256, 256, 0, stream>>>(points, fc_w1, fc_b1, clsC);
    cls_fc2         <<<(B_*16*DCH)/256, 256, 0, stream>>>(clsC, fc_w2, fc_b2, pts_out);
    write_xyz       <<<(B_*(NP+16)*3)/256 + 1, 256, 0, stream>>>(xyz, new_xyz, xyz_out);
}
